// MambaNet_7919919694492
// MI455X (gfx1250) — compile-verified, matrix operand order checked by reading, not run on hardware
//
#include <hip/hip_runtime.h>
#include <hip/hip_bf16.h>

// ---------------------------------------------------------------------------
// MambaNet forward for MI455X (gfx1250, wave32).
// All GEMM-shaped ops use v_wmma_f32_16x16x32_bf16 (f32 accumulate).
// ---------------------------------------------------------------------------

typedef __attribute__((ext_vector_type(16))) __bf16 v16bf;
typedef __attribute__((ext_vector_type(8)))  float  v8f;

union FragB16 {
    v16bf v;
    uint4 q[2];
    unsigned short h[16];
};

__device__ __forceinline__ unsigned short f2bf(float f) {
    unsigned int u = __float_as_uint(f);
    unsigned int r = u + 0x7FFFu + ((u >> 16) & 1u);
    return (unsigned short)(r >> 16);
}

__device__ __forceinline__ float gelu_t(float x) {
    float x3 = x * x * x;
    return 0.5f * x * (1.0f + tanhf(0.7978845608028654f * (x + 0.044715f * x3)));
}
__device__ __forceinline__ float silu_f(float x) {
    return x / (1.0f + __expf(-x));
}

#define PI_INV 0.3183098861837907f

// ============================== feature extraction =========================
// iq (16,2,8192) -> feat (16,5,8192): I,Q,amp,phase,freq
__global__ __launch_bounds__(256) void feat_kern(const float* __restrict__ iq,
                                                 float* __restrict__ feat, int L) {
    int idx = blockIdx.x * 256 + threadIdx.x;
    if (idx >= 16 * L) return;
    int b = idx / L, l = idx % L;
    const float* Ip = iq + (size_t)b * 2 * L;
    float I = Ip[l], Q = Ip[L + l];
    float amp = sqrtf(I * I + Q * Q + 1e-8f);
    float ph = atan2f(Q, I) * PI_INV;
    float freq = 0.f;
    if (l > 0) freq = ph - atan2f(Ip[L + l - 1], Ip[l - 1]) * PI_INV;
    float* F = feat + (size_t)b * 5 * L;
    F[0 * L + l] = I; F[1 * L + l] = Q; F[2 * L + l] = amp;
    F[3 * L + l] = ph; F[4 * L + l] = freq;
}

// ============================== conv stages ================================
__global__ __launch_bounds__(256) void conv1_kern(
    const float* __restrict__ in, const float* __restrict__ w, const float* __restrict__ cb,
    const float* __restrict__ g, const float* __restrict__ bb,
    const float* __restrict__ m, const float* __restrict__ v,
    float* __restrict__ out, int L) {
    int idx = blockIdx.x * 256 + threadIdx.x;
    if (idx >= 16 * 32 * L) return;
    int l = idx % L, co = (idx / L) % 32, b = idx / (32 * L);
    const float* F = in + (size_t)b * 5 * L;
    float acc = cb[co];
    for (int ci = 0; ci < 5; ++ci) {
        const float* wr = w + (co * 5 + ci) * 7;
        const float* fr = F + (size_t)ci * L;
        #pragma unroll
        for (int k = 0; k < 7; ++k) {
            int x = l - 3 + k;
            if (x >= 0 && x < L) acc += wr[k] * fr[x];
        }
    }
    float s = g[co] * rsqrtf(v[co] + 1e-5f);
    out[idx] = gelu_t((acc - m[co]) * s + bb[co]);
}

__global__ __launch_bounds__(256) void conv2_kern(
    const float* __restrict__ in, const float* __restrict__ w, const float* __restrict__ cb,
    const float* __restrict__ g, const float* __restrict__ bb,
    const float* __restrict__ m, const float* __restrict__ v,
    float* __restrict__ out, int L) {
    int idx = blockIdx.x * 256 + threadIdx.x;
    if (idx >= 16 * 64 * L) return;
    int l = idx % L, co = (idx / L) % 64, b = idx / (64 * L);
    const float* F = in + (size_t)b * 32 * L;
    float acc = cb[co];
    for (int ci = 0; ci < 32; ++ci) {
        const float* wr = w + (co * 32 + ci) * 7;
        const float* fr = F + (size_t)ci * L;
        #pragma unroll
        for (int k = 0; k < 7; ++k) {
            int x = l - 3 + k;
            if (x >= 0 && x < L) acc += wr[k] * fr[x];
        }
    }
    float s = g[co] * rsqrtf(v[co] + 1e-5f);
    out[idx] = gelu_t((acc - m[co]) * s + bb[co]);
}

// stride-8 conv, k=8, pad=0, output transposed to (B, 1024, 128)
__global__ __launch_bounds__(256) void conv3_kern(
    const float* __restrict__ in, const float* __restrict__ w, const float* __restrict__ cb,
    const float* __restrict__ g, const float* __restrict__ bb,
    const float* __restrict__ m, const float* __restrict__ v,
    float* __restrict__ out, int Lin) {
    int idx = blockIdx.x * 256 + threadIdx.x;
    if (idx >= 16 * 128 * 1024) return;
    int l = idx % 1024, co = (idx / 1024) % 128, b = idx / (128 * 1024);
    const float* F = in + (size_t)b * 64 * Lin;
    float acc = cb[co];
    for (int ci = 0; ci < 64; ++ci) {
        const float* wr = w + (co * 64 + ci) * 8;
        const float* fr = F + (size_t)ci * Lin + l * 8;
        #pragma unroll
        for (int k = 0; k < 8; ++k) acc += wr[k] * fr[k];
    }
    float s = g[co] * rsqrtf(v[co] + 1e-5f);
    out[((size_t)b * 1024 + l) * 128 + co] = gelu_t((acc - m[co]) * s + bb[co]);
}

// ============================== converts ===================================
__global__ __launch_bounds__(256) void a2bf_kern(const float* __restrict__ src,
                                                 unsigned short* __restrict__ dst, int n) {
    int i = blockIdx.x * 256 + threadIdx.x;
    if (i < n) dst[i] = f2bf(src[i]);
}
// weight (rows x cols) f32 -> (rowsPad x cols) bf16 zero padded
__global__ __launch_bounds__(256) void w2bf_kern(const float* __restrict__ src,
                                                 unsigned short* __restrict__ dst,
                                                 int rows, int cols, int rowsPad) {
    int i = blockIdx.x * 256 + threadIdx.x;
    if (i >= rowsPad * cols) return;
    int r = i / cols;
    dst[i] = (r < rows) ? f2bf(src[i]) : (unsigned short)0;
}

// ============================== generic WMMA GEMM ==========================
// C[M,ldc] (f32) = A[M,K](bf16,row-major) * W[Nw,K](bf16,row-major)^T (+bias)(+acc)
// Block tile 128(M) x 64(N), K step 32. 256 threads = 8 waves; wave w owns
// rows [w*16, w*16+16) and all 4 N sub-tiles.
__global__ __launch_bounds__(256) void gemm_bf16_wmma(
    const unsigned short* __restrict__ A, const unsigned short* __restrict__ W,
    const float* __restrict__ bias, float* __restrict__ C,
    int M, int N, int K, int Nw, int ldc, int flags) {
    __shared__ __align__(16) unsigned short lA[128 * 32];
    __shared__ __align__(16) unsigned short lB[64 * 32];
    const int tid = threadIdx.x;
    const int wave = tid >> 5;
    const int lane = tid & 31;
    const int row16 = lane & 15;
    const int halfk = (lane >> 4) * 8;  // 0 or 8 (CDNA5 16-bit A/B frag layout)
    const int bm = blockIdx.y * 128;
    const int bn = blockIdx.x * 64;

    v8f acc[4];
    #pragma unroll
    for (int i = 0; i < 4; ++i)
        #pragma unroll
        for (int j = 0; j < 8; ++j) acc[i][j] = 0.f;

    for (int k0 = 0; k0 < K; k0 += 32) {
        {   // A tile: 128x32 = 4096 bf16, 16 per thread
            int e = tid * 16;
            int r = e >> 5, c = e & 31;
            const unsigned short* src = A + (size_t)(bm + r) * K + k0 + c;
            *(uint4*)&lA[e]     = *(const uint4*)(src);
            *(uint4*)&lA[e + 8] = *(const uint4*)(src + 8);
            if (k0 + 32 < K) __builtin_prefetch(src + 32, 0, 3); // global_prefetch_b8
        }
        {   // W tile: 64x32 = 2048 bf16, 8 per thread (guard padded rows)
            int e = tid * 8;
            int r = e >> 5, c = e & 31;
            int gn = bn + r;
            uint4 val = make_uint4(0u, 0u, 0u, 0u);
            if (gn < Nw) val = *(const uint4*)(W + (size_t)gn * K + k0 + c);
            *(uint4*)&lB[e] = val;
        }
        __syncthreads();
        FragB16 a;
        {
            const unsigned short* p = &lA[(wave * 16 + row16) * 32];
            a.q[0] = *(const uint4*)(p + halfk);       // K = halfk..halfk+7
            a.q[1] = *(const uint4*)(p + 16 + halfk);  // K = 16+halfk..
        }
        #pragma unroll
        for (int ns = 0; ns < 4; ++ns) {
            FragB16 bfr;
            const unsigned short* p = &lB[(ns * 16 + row16) * 32];
            bfr.q[0] = *(const uint4*)(p + halfk);
            bfr.q[1] = *(const uint4*)(p + 16 + halfk);
            acc[ns] = __builtin_amdgcn_wmma_f32_16x16x32_bf16(
                false, a.v, false, bfr.v, (short)0, acc[ns], false, false);
        }
        __syncthreads();
    }
    // D layout: lane = N col (lane&15), vgpr r -> M row = r + 8*(lane>>4)
    int mrow = bm + wave * 16 + (lane >> 4) * 8;
    #pragma unroll
    for (int ns = 0; ns < 4; ++ns) {
        int col = bn + ns * 16 + row16;
        if (col < N) {
            float bval = (flags & 1) ? bias[col] : 0.f;
            #pragma unroll
            for (int r = 0; r < 8; ++r) {
                size_t off = (size_t)(mrow + r) * ldc + col;
                float val = acc[ns][r] + bval;
                if (flags & 2) val += C[off];
                C[off] = val;
            }
        }
    }
}

// ============================== attention ==================================
// Flash-style attention, hd=16 (K padded to 32). grid (L/128, H=8, B=16),
// 256 threads = 8 waves; each wave owns 16 query rows. Output written as bf16.
__global__ __launch_bounds__(256) void attn_kern(const float* __restrict__ qkv,
                                                 unsigned short* __restrict__ out, int L) {
    __shared__ __align__(16) unsigned short lK[16 * 32];   // keys x hd(padded)
    __shared__ __align__(16) unsigned short lVT[16 * 16];  // hd x keys (transposed)
    __shared__ __align__(16) unsigned short lP[8 * 256];   // per-wave P tile (16x16)
    const int tid = threadIdx.x;
    const int wave = tid >> 5, lane = tid & 31;
    const int row16 = lane & 15;
    const int halfk = (lane >> 4) * 8;
    const int rt = blockIdx.x, h = blockIdx.y, b = blockIdx.z;
    const size_t bL = (size_t)b * L;
    const int m0 = rt * 128 + wave * 16;

    // zero K tile once (pad columns 16..31 stay zero forever)
    lK[tid] = 0; lK[tid + 256] = 0;
    __syncthreads();

    // Q fragment: rows m0+row16, K=halfk.., scaled by 1/sqrt(16)
    FragB16 qa;
    {
        const float* qp = qkv + (bL + m0 + row16) * 384 + h * 16 + halfk;
        #pragma unroll
        for (int i = 0; i < 8; ++i) qa.h[i] = f2bf(qp[i] * 0.25f);
        #pragma unroll
        for (int i = 8; i < 16; ++i) qa.h[i] = 0;
    }
    float mi[8], li[8];
    v8f o;
    #pragma unroll
    for (int r = 0; r < 8; ++r) { mi[r] = -1e30f; li[r] = 0.f; o[r] = 0.f; }

    for (int j = 0; j < L / 16; ++j) {
        __syncthreads();  // previous iteration consumers done
        {   // load K/V 16-key tile (one element per thread)
            int key = tid >> 4, d = tid & 15;
            const float* base = qkv + (bL + (size_t)j * 16 + key) * 384 + h * 16 + d;
            lK[key * 32 + d]  = f2bf(base[128]);
            lVT[d * 16 + key] = f2bf(base[256]);
        }
        __syncthreads();
        // S = Q * K^T  (one WMMA)
        v8f S;
        #pragma unroll
        for (int r = 0; r < 8; ++r) S[r] = 0.f;
        {
            FragB16 kb;
            const unsigned short* p = &lK[row16 * 32];
            kb.q[0] = *(const uint4*)(p + halfk);
            kb.q[1] = *(const uint4*)(p + 16 + halfk);   // zeros (hd pad)
            S = __builtin_amdgcn_wmma_f32_16x16x32_bf16(
                false, qa.v, false, kb.v, (short)0, S, false, false);
        }
        // online softmax per row (half-wave reductions over 16 lanes)
        #pragma unroll
        for (int r = 0; r < 8; ++r) {
            float s = S[r];
            float mx = s;
            #pragma unroll
            for (int d = 1; d < 16; d <<= 1) mx = fmaxf(mx, __shfl_xor(mx, d, 16));
            float mn = fmaxf(mi[r], mx);
            float scale = __expf(mi[r] - mn);
            float p = __expf(s - mn);
            float ps = p;
            #pragma unroll
            for (int d = 1; d < 16; d <<= 1) ps += __shfl_xor(ps, d, 16);
            li[r] = li[r] * scale + ps;
            mi[r] = mn;
            o[r] *= scale;
            int m = r + ((lane >> 4) << 3);
            lP[wave * 256 + m * 16 + row16] = f2bf(p);
        }
        __syncthreads();  // make lP visible (C-layout -> A-layout transpose)
        // O += P * V (one WMMA)
        {
            FragB16 pa, vb;
            pa.q[0] = *(const uint4*)&lP[wave * 256 + row16 * 16 + halfk];
            pa.q[1] = make_uint4(0u, 0u, 0u, 0u);   // K pad (only 16 keys)
            vb.q[0] = *(const uint4*)&lVT[row16 * 16 + halfk];
            vb.q[1] = make_uint4(0u, 0u, 0u, 0u);
            o = __builtin_amdgcn_wmma_f32_16x16x32_bf16(
                false, pa.v, false, vb.v, (short)0, o, false, false);
        }
    }
    // normalize + store bf16 (feeds the wo GEMM directly)
    #pragma unroll
    for (int r = 0; r < 8; ++r) {
        int row = m0 + r + ((lane >> 4) << 3);
        out[(bL + row) * 128 + h * 16 + row16] = f2bf(o[r] / li[r]);
    }
}

// ============================== add + LayerNorm (+FiLM) ====================
__global__ __launch_bounds__(128) void add_ln_kern(
    const float* __restrict__ x, const float* __restrict__ r,
    const float* __restrict__ g, const float* __restrict__ bta,
    float* __restrict__ out, unsigned short* __restrict__ outbf,
    const float* __restrict__ film_w, const float* __restrict__ film_b,
    const float* __restrict__ snr_db) {
    int row = blockIdx.x;
    int c = threadIdx.x;
    __shared__ float red[128];
    size_t off = (size_t)row * 128 + c;
    float val = x[off] + r[off];
    red[c] = val; __syncthreads();
    for (int s = 64; s > 0; s >>= 1) { if (c < s) red[c] += red[c + s]; __syncthreads(); }
    float mean = red[0] * (1.f / 128.f); __syncthreads();
    float d = val - mean;
    red[c] = d * d; __syncthreads();
    for (int s = 64; s > 0; s >>= 1) { if (c < s) red[c] += red[c + s]; __syncthreads(); }
    float var = red[0] * (1.f / 128.f);
    float y = d * rsqrtf(var + 1e-5f) * g[c] + bta[c];
    if (film_w) {
        int b = row >> 10;
        float sn = (snr_db[b] + 10.f) * (1.f / 40.f);
        float gamma = film_w[c] * sn + film_b[c];
        float beta  = film_w[128 + c] * sn + film_b[128 + c];
        y = y * (1.f + gamma) + beta;
    }
    out[off] = y;
    if (outbf) outbf[off] = f2bf(y);
}

// ============================== Mamba pieces ===============================
// dt = softplus(zx[:,640+hd] + dt_bias); dA = exp(-exp(A_log)*dt)
__global__ __launch_bounds__(256) void dtda_kern(const float* __restrict__ zx,
                                                 const float* __restrict__ dt_bias,
                                                 const float* __restrict__ A_log,
                                                 float* __restrict__ dt, float* __restrict__ dA) {
    int i = blockIdx.x * 256 + threadIdx.x;
    if (i >= 16 * 1024 * 4) return;
    int hd = i & 3, row = i >> 2;
    float xx = zx[(size_t)row * 656 + 640 + hd] + dt_bias[hd];
    float sp = (xx > 20.f) ? xx : log1pf(__expf(xx));
    dt[i] = sp;
    dA[i] = __expf(-__expf(A_log[hd]) * sp);
}

// depthwise causal conv (k=4) + silu; dir=1 flips the sequence sense
__global__ __launch_bounds__(256) void convdw_kern(const float* __restrict__ zx,
                                                   const float* __restrict__ w,
                                                   const float* __restrict__ cb,
                                                   float* __restrict__ out, int dir) {
    int i = blockIdx.x * 256 + threadIdx.x;
    if (i >= 16 * 1024 * 384) return;
    int c = i % 384, l = (i / 384) % 1024, b = i / (384 * 1024);
    float acc = cb[c];
    #pragma unroll
    for (int k = 0; k < 4; ++k) {
        int idx = dir ? (l + 3 - k) : (l - 3 + k);
        if (idx >= 0 && idx < 1024)
            acc += w[c * 4 + k] * zx[((size_t)b * 1024 + idx) * 656 + 256 + c];
    }
    out[i] = silu_f(acc);
}

// selective scan: grid 64 = (b, head); thread t: p = t&63, group g = t>>6
// holds 16 DSTATE entries of h[p, :] in registers; LDS-reduce y over groups.
__global__ __launch_bounds__(256) void scan_kern(const float* __restrict__ xBCc,
                                                 const float* __restrict__ dtb,
                                                 const float* __restrict__ dAb,
                                                 const float* __restrict__ Dp,
                                                 float* __restrict__ y, int dir) {
    int b = blockIdx.x >> 2, hd = blockIdx.x & 3;
    int t = threadIdx.x, p = t & 63, g = t >> 6;
    float h[16];
    #pragma unroll
    for (int j = 0; j < 16; ++j) h[j] = 0.f;
    __shared__ float part[256];
    float Dv = Dp[hd];
    for (int l = 0; l < 1024; ++l) {
        int li = dir ? (1023 - l) : l;
        size_t base = (size_t)b * 1024 + li;
        float dA = dAb[base * 4 + hd];
        float dt = dtb[base * 4 + hd];
        const float* xr = xBCc + base * 384;
        float coef = dt * xr[hd * 64 + p];
        const float* Br = xr + 256 + g * 16;
        const float* Cr = xr + 320 + g * 16;
        float ysum = 0.f;
        #pragma unroll
        for (int j = 0; j < 16; ++j) {
            h[j] = h[j] * dA + coef * Br[j];
            ysum += h[j] * Cr[j];
        }
        part[t] = ysum;
        __syncthreads();
        if (t < 64) {
            float yy = part[t] + part[t + 64] + part[t + 128] + part[t + 192]
                     + Dv * xr[hd * 64 + t];
            y[base * 256 + hd * 64 + t] = yy;
        }
        __syncthreads();
    }
}

// y = rmsnorm(y_scan * silu(z)) * norm_w  -> bf16 (feeds out_proj GEMM)
__global__ __launch_bounds__(256) void gate_rms_kern(const float* __restrict__ yscan,
                                                     const float* __restrict__ zx,
                                                     const float* __restrict__ nw,
                                                     unsigned short* __restrict__ ybf) {
    int row = blockIdx.x, c = threadIdx.x;
    __shared__ float red[256];
    float z = zx[(size_t)row * 656 + c];
    float yv = yscan[(size_t)row * 256 + c] * silu_f(z);
    red[c] = yv * yv; __syncthreads();
    for (int s = 128; s > 0; s >>= 1) { if (c < s) red[c] += red[c + s]; __syncthreads(); }
    float ms = red[0] * (1.f / 256.f);
    ybf[(size_t)row * 256 + c] = f2bf(yv * rsqrtf(ms + 1e-5f) * nw[c]);
}

// ============================== heads ======================================
__global__ __launch_bounds__(128) void bits_kern(const float* __restrict__ hf,
                                                 const float* __restrict__ bw,
                                                 const float* __restrict__ bb,
                                                 float* __restrict__ out) {
    int row = blockIdx.x, c = threadIdx.x;
    __shared__ float red[128];
    red[c] = hf[(size_t)row * 128 + c] * bw[c]; __syncthreads();
    for (int s = 64; s > 0; s >>= 1) { if (c < s) red[c] += red[c + s]; __syncthreads(); }
    if (c == 0) out[row] = red[0] + bb[0];
}

__global__ __launch_bounds__(128) void snr_kern(const float* __restrict__ hf,
                                                const float* __restrict__ sw,
                                                const float* __restrict__ sb,
                                                float* __restrict__ out) {
    int b = blockIdx.x, c = threadIdx.x;
    float s = 0.f;
    for (int l = 0; l < 1024; ++l) s += hf[((size_t)b * 1024 + l) * 128 + c];
    __shared__ float red[128];
    red[c] = (s * (1.f / 1024.f)) * sw[c]; __syncthreads();
    for (int st = 64; st > 0; st >>= 1) { if (c < st) red[c] += red[c + st]; __syncthreads(); }
    if (c == 0) out[b] = red[0] + sb[0];
}

// ============================== host launch ================================
static inline int cdiv(int a, int b) { return (a + b - 1) / b; }

extern "C" void kernel_launch(void* const* d_in, const int* in_sizes, int n_in,
                              void* d_out, int out_size, void* d_ws, size_t ws_size,
                              hipStream_t stream) {
    (void)in_sizes; (void)n_in; (void)out_size; (void)ws_size;
    const float* iq      = (const float*)d_in[0];
    const float* snr_db  = (const float*)d_in[1];
    const float* c1w = (const float*)d_in[2],  *c1b = (const float*)d_in[3];
    const float* bn1g=(const float*)d_in[4], *bn1b=(const float*)d_in[5],
               * bn1m=(const float*)d_in[6], *bn1v=(const float*)d_in[7];
    const float* c2w = (const float*)d_in[8],  *c2b = (const float*)d_in[9];
    const float* bn2g=(const float*)d_in[10], *bn2b=(const float*)d_in[11],
               * bn2m=(const float*)d_in[12], *bn2v=(const float*)d_in[13];
    const float* c3w = (const float*)d_in[14], *c3b = (const float*)d_in[15];
    const float* bn3g=(const float*)d_in[16], *bn3b=(const float*)d_in[17],
               * bn3m=(const float*)d_in[18], *bn3v=(const float*)d_in[19];
    const float* wqkv=(const float*)d_in[20], *bqkv=(const float*)d_in[21];
    const float* wo  =(const float*)d_in[22], *bo  =(const float*)d_in[23];
    const float* ln1g=(const float*)d_in[24], *ln1b=(const float*)d_in[25];
    const float* ln2g=(const float*)d_in[26], *ln2b=(const float*)d_in[27];
    // mamba params: fwd base 28, bwd base 36 (in_proj, conv_w, conv_b, dt_bias,
    //                                          A_log, D, norm_w, out_proj)
    const float* film_w=(const float*)d_in[44], *film_b=(const float*)d_in[45];
    const float* bit_w =(const float*)d_in[46], *bit_b =(const float*)d_in[47];
    const float* snr_w =(const float*)d_in[48], *snr_b =(const float*)d_in[49];

    char* ws = (char*)d_ws;
    constexpr size_t MB = 1ull << 20;
    // phase A
    float* feat  = (float*)(ws + 0);
    float* h1    = (float*)(ws + 4  * MB);
    float* h2    = (float*)(ws + 24 * MB);
    float* h3    = (float*)(ws + 60 * MB);
    unsigned short* h3bf = (unsigned short*)(ws + 70 * MB);
    // phase B (reuses [0..58MB))
    float* qkv   = (float*)(ws + 0);
    unsigned short* attnbf = (unsigned short*)(ws + 26 * MB);
    float* attnO = (float*)(ws + 31 * MB);
    float* hln1  = (float*)(ws + 75 * MB);
    unsigned short* hln1bf = (unsigned short*)(ws + 84 * MB);
    float* bi    = (float*)(ws + 89 * MB);
    // phase C (reuses [0..70MB))
    float* zx    = (float*)(ws + 0);         // 16*1024*656 f32
    float* xbc   = (float*)(ws + 44 * MB);   // 16*1024*384 f32
    float* dtbuf = (float*)(ws + 98 * MB);
    float* dAbuf = (float*)(ws + 99 * MB);
    float* yscan = (float*)(ws + 100 * MB);
    unsigned short* ybf = (unsigned short*)(ws + 117 * MB);
    float* hfin  = (float*)(ws + 0);         // phase D
    // bf16 weights
    unsigned short* wqkvbf = (unsigned short*)(ws + 126 * MB);
    unsigned short* wobf   = (unsigned short*)(ws + 127 * MB);
    unsigned short* wipf   = (unsigned short*)(ws + 128 * MB);
    unsigned short* wipb   = (unsigned short*)(ws + 129 * MB);
    unsigned short* wopf   = (unsigned short*)(ws + 130 * MB);
    unsigned short* wopb   = (unsigned short*)(ws + 131 * MB);

    const int Lin = 8192, L = 1024, Mrows = 16 * 1024;

    // features + conv stack
    feat_kern <<<cdiv(16 * Lin, 256), 256, 0, stream>>>(iq, feat, Lin);
    conv1_kern<<<cdiv(16 * 32 * Lin, 256), 256, 0, stream>>>(feat, c1w, c1b, bn1g, bn1b, bn1m, bn1v, h1, Lin);
    conv2_kern<<<cdiv(16 * 64 * Lin, 256), 256, 0, stream>>>(h1, c2w, c2b, bn2g, bn2b, bn2m, bn2v, h2, Lin);
    conv3_kern<<<cdiv(16 * 128 * L, 256), 256, 0, stream>>>(h2, c3w, c3b, bn3g, bn3b, bn3m, bn3v, h3, Lin);

    // weight conversions
    a2bf_kern<<<cdiv(Mrows * 128, 256), 256, 0, stream>>>(h3, h3bf, Mrows * 128);
    w2bf_kern<<<cdiv(384 * 128, 256), 256, 0, stream>>>(wqkv, wqkvbf, 384, 128, 384);
    w2bf_kern<<<cdiv(128 * 128, 256), 256, 0, stream>>>(wo, wobf, 128, 128, 128);
    w2bf_kern<<<cdiv(656 * 128, 256), 256, 0, stream>>>((const float*)d_in[28], wipf, 644, 128, 656);
    w2bf_kern<<<cdiv(656 * 128, 256), 256, 0, stream>>>((const float*)d_in[36], wipb, 644, 128, 656);
    w2bf_kern<<<cdiv(128 * 256, 256), 256, 0, stream>>>((const float*)d_in[35], wopf, 128, 256, 128);
    w2bf_kern<<<cdiv(128 * 256, 256), 256, 0, stream>>>((const float*)d_in[43], wopb, 128, 256, 128);

    // attention
    gemm_bf16_wmma<<<dim3(6, 128), 256, 0, stream>>>(h3bf, wqkvbf, bqkv, qkv,
                                                     Mrows, 384, 128, 384, 384, 1);
    attn_kern<<<dim3(L / 128, 8, 16), 256, 0, stream>>>(qkv, attnbf, L);
    gemm_bf16_wmma<<<dim3(2, 128), 256, 0, stream>>>(attnbf, wobf, bo, attnO,
                                                     Mrows, 128, 128, 128, 128, 1);
    add_ln_kern<<<Mrows, 128, 0, stream>>>(h3, attnO, ln1g, ln1b, hln1, hln1bf,
                                           nullptr, nullptr, nullptr);

    // bidirectional mamba2
    for (int dir = 0; dir < 2; ++dir) {
        int base = dir ? 36 : 28;
        const unsigned short* wip = dir ? wipb : wipf;
        const unsigned short* wop = dir ? wopb : wopf;
        gemm_bf16_wmma<<<dim3(11, 128), 256, 0, stream>>>(hln1bf, wip, nullptr, zx,
                                                          Mrows, 644, 128, 656, 656, 0);
        dtda_kern<<<cdiv(Mrows * 4, 256), 256, 0, stream>>>(
            zx, (const float*)d_in[base + 3], (const float*)d_in[base + 4], dtbuf, dAbuf);
        convdw_kern<<<cdiv(Mrows * 384, 256), 256, 0, stream>>>(
            zx, (const float*)d_in[base + 1], (const float*)d_in[base + 2], xbc, dir);
        scan_kern<<<64, 256, 0, stream>>>(xbc, dtbuf, dAbuf,
                                          (const float*)d_in[base + 5], yscan, dir);
        gate_rms_kern<<<Mrows, 256, 0, stream>>>(yscan, zx, (const float*)d_in[base + 6], ybf);
        gemm_bf16_wmma<<<dim3(2, 128), 256, 0, stream>>>(ybf, wop, nullptr, bi,
                                                         Mrows, 128, 256, 128, 128, dir ? 2 : 0);
    }

    // ln2 + FiLM, heads
    add_ln_kern<<<Mrows, 128, 0, stream>>>(hln1, bi, ln2g, ln2b, hfin, nullptr,
                                           film_w, film_b, snr_db);
    float* out = (float*)d_out;
    bits_kern<<<Mrows, 128, 0, stream>>>(hfin, bit_w, bit_b, out);
    snr_kern<<<16, 128, 0, stream>>>(hfin, snr_w, snr_b, out + Mrows);
}
